// CCGQAAttention_59657095741552
// MI455X (gfx1250) — compile-verified, matrix operand order checked by reading, not run on hardware
//
#include <hip/hip_runtime.h>
#include <stdint.h>

// ---------------------------------------------------------------- constants
#define NHEADS   16
#define NKV      4
#define HDIM     64
#define DMODEL   4096
#define LATENT   1024          // NHEADS*HDIM
#define KVD      256           // NKV*HDIM
#define QKVOUT   1536          // LATENT + 2*KVD
#define SEQ      2048
#define NBATCH   2
#define TOK      (NBATCH*SEQ)  // 4096

typedef __bf16 bf16_t;
typedef __attribute__((ext_vector_type(16))) __bf16 v16bf;
typedef __attribute__((ext_vector_type(8)))  __bf16 v8bf;
typedef __attribute__((ext_vector_type(8)))  float  v8f;

__device__ __forceinline__ bf16_t f2bf(float f) {
  uint32_t u = __builtin_bit_cast(uint32_t, f);
  uint32_t r = u + 0x7FFFu + ((u >> 16) & 1u);   // round-to-nearest-even
  uint16_t h = (uint16_t)(r >> 16);
  return __builtin_bit_cast(bf16_t, h);
}

__device__ __forceinline__ v8f zero_v8f() {
  v8f z;
#pragma unroll
  for (int i = 0; i < 8; ++i) z[i] = 0.f;
  return z;
}

// pack two 8-element halves into the 16-bit WMMA fragment register layout
__device__ __forceinline__ v16bf pack16(v8bf lo, v8bf hi) {
  v16bf r;
#pragma unroll
  for (int j = 0; j < 8; ++j) { r[j] = lo[j]; r[8 + j] = hi[j]; }
  return r;
}

// table-order K fragment load: lane points at K=koff, second half at K=koff+16
__device__ __forceinline__ v16bf load_frag16(const bf16_t* p) {
  v8bf lo = *(const v8bf*)p;
  v8bf hi = *(const v8bf*)(p + 16);
  return pack16(lo, hi);
}

// CDNA5 async copy: 16B global -> LDS per lane, tracked by ASYNCcnt.
// VDST = LDS byte address (low 32 bits of generic pointer), VADDR = 64b addr.
__device__ __forceinline__ void async_cp_b128(const bf16_t* g, bf16_t* l) {
  const uint32_t lds = (uint32_t)(uintptr_t)l;
  asm volatile("global_load_async_to_lds_b128 %0, %1, off"
               :: "v"(lds), "v"(g) : "memory");
}

// ---------------------------------------------------------------- fp32->bf16
__global__ void k_f32_to_bf16(const float* __restrict__ src,
                              bf16_t* __restrict__ dst, int n) {
  int i = blockIdx.x * blockDim.x + threadIdx.x;
  if (i < n) dst[i] = f2bf(src[i]);
}

// ---------------------------------------------------------------- bf16 GEMM
// C[M,N] (f32) = A[M,K](bf16, row-major) * B[N,K](bf16, row-major)^T
// block tile 128x128x32, 256 threads = 8 waves (2 M-waves x 4 N-waves),
// wave tile 64x32 = 4x2 wmma tiles. M,N,K must be multiples of 128/128/32.
// LDS is double-buffered; tiles are fetched with global_load_async_to_lds_b128
// one K-step ahead (s_wait_asynccnt 4 keeps next buffer's copies in flight),
// and global_prefetch_b8 warms L2 one further K-step out.
#define BM 128
#define BN 128
#define BKK 32
#define LDT 40   // padded LDS row stride (bf16): 80B, 16B aligned

__global__ __launch_bounds__(256) void k_gemm_bf16_nt(
    const bf16_t* __restrict__ A, const bf16_t* __restrict__ B,
    float* __restrict__ C, int M, int N, int K) {
  __shared__ bf16_t As[2][BM * LDT];
  __shared__ bf16_t Bs[2][BN * LDT];

  const int tid  = threadIdx.x;
  const int lane = tid & 31;
  const int wid  = tid >> 5;
  const int wm   = wid & 1;          // 0..1
  const int wn   = wid >> 1;         // 0..3
  const int bm0  = blockIdx.y * BM;
  const int bn0  = blockIdx.x * BN;

  // staging: each thread owns one 16-element segment of A and of B per K-step
  const int srow  = tid >> 1;        // 0..127
  const int shalf = (tid & 1) * 16;  // 0 or 16
  const int ln    = lane & 15;
  const int koff  = (lane >> 4) * 8;

  const bf16_t* ga0 = A + (size_t)(bm0 + srow) * K + shalf;
  const bf16_t* gb0 = B + (size_t)(bn0 + srow) * K + shalf;
  const int sdst = srow * LDT + shalf;

  v8f acc[4][2];
#pragma unroll
  for (int mt = 0; mt < 4; ++mt)
#pragma unroll
    for (int nt = 0; nt < 2; ++nt) acc[mt][nt] = zero_v8f();

  // prologue: stage first K-tile into buffer 0 (4 async b128 per thread)
  async_cp_b128(ga0,     &As[0][sdst]);
  async_cp_b128(ga0 + 8, &As[0][sdst + 8]);
  async_cp_b128(gb0,     &Bs[0][sdst]);
  async_cp_b128(gb0 + 8, &Bs[0][sdst + 8]);

  int cur = 0;
  for (int k0 = 0; k0 < K; k0 += BKK) {
    const bool more = (k0 + BKK) < K;
    if (more) {
      const bf16_t* ga = ga0 + k0 + BKK;
      const bf16_t* gb = gb0 + k0 + BKK;
      __builtin_prefetch(ga + BKK, 0, 1);          // L2 warm, 2 tiles ahead
      __builtin_prefetch(gb + BKK, 0, 1);
      async_cp_b128(ga,     &As[cur ^ 1][sdst]);   // next tile, other buffer
      async_cp_b128(ga + 8, &As[cur ^ 1][sdst + 8]);
      async_cp_b128(gb,     &Bs[cur ^ 1][sdst]);
      async_cp_b128(gb + 8, &Bs[cur ^ 1][sdst + 8]);
      // async loads complete in order: <=4 outstanding means the 4 copies
      // for the *current* buffer have landed; next buffer's stay in flight.
      asm volatile("s_wait_asynccnt 0x4" ::: "memory");
    } else {
      asm volatile("s_wait_asynccnt 0x0" ::: "memory");
    }
    __syncthreads();   // cross-wave visibility of current buffer

    v16bf af[4], bfr[2];
#pragma unroll
    for (int mt = 0; mt < 4; ++mt)
      af[mt] = load_frag16(&As[cur][(wm * 64 + mt * 16 + ln) * LDT + koff]);
#pragma unroll
    for (int nt = 0; nt < 2; ++nt)
      bfr[nt] = load_frag16(&Bs[cur][(wn * 32 + nt * 16 + ln) * LDT + koff]);

#pragma unroll
    for (int mt = 0; mt < 4; ++mt)
#pragma unroll
      for (int nt = 0; nt < 2; ++nt)
        acc[mt][nt] = __builtin_amdgcn_wmma_f32_16x16x32_bf16(
            false, af[mt], false, bfr[nt], (short)0, acc[mt][nt], false, false);
    __syncthreads();   // all waves done reading before buffer is re-staged
    cur ^= 1;
  }

  const int rbase = (lane >> 4) * 8;   // C layout: VGPR r -> row r (+8 hi half)
#pragma unroll
  for (int mt = 0; mt < 4; ++mt)
#pragma unroll
    for (int nt = 0; nt < 2; ++nt) {
      const int row0 = bm0 + wm * 64 + mt * 16 + rbase;
      const int col  = bn0 + wn * 32 + nt * 16 + ln;
#pragma unroll
      for (int r = 0; r < 8; ++r)
        C[(size_t)(row0 + r) * N + col] = acc[mt][nt][r];
    }
}

// ---------------------------------------------------- grouped causal conv (k=3)
// Y[t,o] = b[o] + sum_dt sum_ci W[o,ci,2-dt] * X[t-dt, col0 + gbase+ci]
__global__ void k_grouped_causal_conv(const float* __restrict__ X, int ldx, int col0,
                                      const float* __restrict__ W,
                                      const float* __restrict__ Bb,
                                      float* __restrict__ Y, int C, int Cin) {
  int i = blockIdx.x * blockDim.x + threadIdx.x;
  if (i >= TOK * C) return;
  const int t = i / C, o = i - t * C;
  const int s = t & (SEQ - 1);
  const int gbase = (o / Cin) * Cin;
  float acc = Bb[o];
#pragma unroll
  for (int dt = 0; dt < 3; ++dt) {
    if (s < dt) continue;
    const float* xr = X + (size_t)(t - dt) * ldx + col0 + gbase;
    const float* wr = W + (size_t)o * (Cin * 3) + (2 - dt);
    float a = 0.f;
    for (int c = 0; c < Cin; ++c) a += wr[c * 3] * xr[c];
    acc += a;
  }
  Y[(size_t)t * C + o] = acc;
}

// ---------------------------------------------- qc2 conv as GEMM: build operands
// Xc[t, dt*C + c] = Y1[t-dt, c]  (causal, per-sequence), bf16
__global__ void k_build_taps(const float* __restrict__ Y1,
                             bf16_t* __restrict__ Xc, int C) {
  int i = blockIdx.x * blockDim.x + threadIdx.x;
  if (i >= TOK * 3 * C) return;
  const int t = i / (3 * C);
  const int rem = i - t * 3 * C;
  const int dt = rem / C;
  const int c = rem - dt * C;
  const int s = t & (SEQ - 1);
  const float v = (s >= dt) ? Y1[(size_t)(t - dt) * C + c] : 0.f;
  Xc[i] = f2bf(v);
}

// Wc[o, dt*C + c] = W[o, c, 2-dt], bf16
__global__ void k_build_wc(const float* __restrict__ W,
                           bf16_t* __restrict__ Wc, int O, int C) {
  int i = blockIdx.x * blockDim.x + threadIdx.x;
  if (i >= O * 3 * C) return;
  const int o = i / (3 * C);
  const int rem = i - o * 3 * C;
  const int dt = rem / C;
  const int c = rem - dt * C;
  Wc[i] = f2bf(W[((size_t)o * C + c) * 3 + (2 - dt)]);
}

// ---------------------------------------------------------------- kc2 (small)
__global__ void k_kconv2(const float* __restrict__ K1, const float* __restrict__ W,
                         const float* __restrict__ Bb, float* __restrict__ K2) {
  int i = blockIdx.x * blockDim.x + threadIdx.x;
  if (i >= TOK * KVD) return;
  const int t = i / KVD, o = i - t * KVD;
  const int s = t & (SEQ - 1);
  float acc = Bb[o];
#pragma unroll
  for (int dt = 0; dt < 3; ++dt) {
    if (s < dt) continue;
    const float* xr = K1 + (size_t)(t - dt) * KVD;
    const float* wr = W + (size_t)o * (KVD * 3) + (2 - dt);
    float a = 0.f;
    for (int c = 0; c < KVD; ++c) a += wr[c * 3] * xr[c];
    acc += a;
  }
  K2[i] = acc;
}

// ------------------------------------------------ finalize Q: mix+norm+rope
// one wave per (token, head); lane handles dim pair (2*lane, 2*lane+1)
__global__ __launch_bounds__(256) void k_finalize_q(
    const float* __restrict__ qkv, const float* __restrict__ Q2,
    const float* __restrict__ q2b, bf16_t* __restrict__ Qo) {
  const int gw = (blockIdx.x * blockDim.x + threadIdx.x) >> 5;
  const int lane = threadIdx.x & 31;
  if (gw >= TOK * NHEADS) return;
  const int t = gw / NHEADS, h = gw - t * NHEADS;
  const int s = t & (SEQ - 1), b = t >> 11;
  const int d0 = lane * 2, d1 = d0 + 1;

  float km0 = 0.f, km1 = 0.f;                       // k_mean over pre-conv K
#pragma unroll
  for (int g = 0; g < NKV; ++g) {
    km0 += qkv[(size_t)t * QKVOUT + LATENT + g * HDIM + d0];
    km1 += qkv[(size_t)t * QKVOUT + LATENT + g * HDIM + d1];
  }
  km0 *= 0.25f; km1 *= 0.25f;

  const int c0 = h * HDIM + d0;
  float q0 = Q2[(size_t)t * LATENT + c0]     + q2b[c0]     + 0.5f * km0;
  float q1 = Q2[(size_t)t * LATENT + c0 + 1] + q2b[c0 + 1] + 0.5f * km1;

  float n2 = q0 * q0 + q1 * q1;
#pragma unroll
  for (int off = 1; off < 32; off <<= 1) n2 += __shfl_xor(n2, off, 32);
  const float inv = 1.0f / fmaxf(sqrtf(n2), 1e-12f);
  q0 *= inv; q1 *= inv;

  const float freq = __powf(10000.0f, -(float)d0 * (1.0f / HDIM));
  float sn, cs; __sincosf((float)s * freq, &sn, &cs);
  const float scale = 0.125f;                       // HDIM^-0.5 folded into Q
  const size_t ob = (((size_t)(b * NHEADS + h)) * SEQ + s) * HDIM;
  Qo[ob + d0] = f2bf((q0 * cs - q1 * sn) * scale);
  Qo[ob + d1] = f2bf((q0 * sn + q1 * cs) * scale);
}

// ------------------------------------------ finalize K (temp folded) + V shift
__global__ __launch_bounds__(256) void k_finalize_kv(
    const float* __restrict__ qkv, const float* __restrict__ K2,
    const float* __restrict__ k2b, const float* __restrict__ key_temp,
    bf16_t* __restrict__ Ko, bf16_t* __restrict__ Vo) {
  const int gw = (blockIdx.x * blockDim.x + threadIdx.x) >> 5;
  const int lane = threadIdx.x & 31;
  if (gw >= TOK * NKV) return;
  const int t = gw / NKV, g = gw - t * NKV;
  const int s = t & (SEQ - 1), b = t >> 11;
  const int d0 = lane * 2, d1 = d0 + 1;

  float qm0 = 0.f, qm1 = 0.f;                       // q_mean over pre-conv Q
#pragma unroll
  for (int h = 0; h < NHEADS; ++h) {
    qm0 += qkv[(size_t)t * QKVOUT + h * HDIM + d0];
    qm1 += qkv[(size_t)t * QKVOUT + h * HDIM + d1];
  }
  qm0 *= (1.f / 16.f); qm1 *= (1.f / 16.f);

  const int c0 = g * HDIM + d0;
  float k0 = K2[(size_t)t * KVD + c0]     + k2b[c0]     + 0.5f * qm0;
  float k1 = K2[(size_t)t * KVD + c0 + 1] + k2b[c0 + 1] + 0.5f * qm1;

  float n2 = k0 * k0 + k1 * k1;
#pragma unroll
  for (int off = 1; off < 32; off <<= 1) n2 += __shfl_xor(n2, off, 32);
  const float inv = key_temp[0] / fmaxf(sqrtf(n2), 1e-12f);
  k0 *= inv; k1 *= inv;

  const float freq = __powf(10000.0f, -(float)d0 * (1.0f / HDIM));
  float sn, cs; __sincosf((float)s * freq, &sn, &cs);
  const size_t ob = (((size_t)(b * NKV + g)) * SEQ + s) * HDIM;
  Ko[ob + d0] = f2bf(k0 * cs - k1 * sn);
  Ko[ob + d1] = f2bf(k0 * sn + k1 * cs);

  // V: channels [0,128) current token, [128,256) previous token
#pragma unroll
  for (int e = 0; e < 2; ++e) {
    const int d = d0 + e, dd = g * HDIM + d;
    float v;
    if (dd < 128)      v = qkv[(size_t)t * QKVOUT + 1280 + dd];
    else if (s > 0)    v = qkv[(size_t)(t - 1) * QKVOUT + 1280 + dd];
    else               v = 0.f;
    Vo[ob + d] = f2bf(v);
  }
}

// ------------------------------------------------------- flash attention (WMMA)
// grid (SEQ/128, NHEADS, NBATCH), 256 threads; wave owns 16 query rows,
// streams keys in chunks of 32, online softmax, O accumulated 16x64.
__global__ __launch_bounds__(256) void k_flash_attn(
    const bf16_t* __restrict__ Q, const bf16_t* __restrict__ Kb,
    const bf16_t* __restrict__ Vb, bf16_t* __restrict__ O) {
  __shared__ bf16_t Pds[8][16 * LDT];   // wave-private P staging (C->A relayout)

  const int lane = threadIdx.x & 31;
  const int wid  = threadIdx.x >> 5;
  const int qt = blockIdx.x, h = blockIdx.y, b = blockIdx.z;
  const int g  = h / (NHEADS / NKV);
  const int qi0 = qt * 128 + wid * 16;
  const int ln = lane & 15;
  const int koff  = (lane >> 4) * 8;
  const int rbase = (lane >> 4) * 8;

  const size_t qbase = ((size_t)(b * NHEADS + h)) * SEQ * HDIM;
  const size_t kvb   = ((size_t)(b * NKV + g)) * SEQ * HDIM;

  v16bf aq[2];
#pragma unroll
  for (int f = 0; f < 2; ++f)
    aq[f] = load_frag16(Q + qbase + (size_t)(qi0 + ln) * HDIM + f * 32 + koff);

  v8f vacc[4];
  float rmax[8], rsum[8], alpha[8];
#pragma unroll
  for (int d4 = 0; d4 < 4; ++d4) vacc[d4] = zero_v8f();
#pragma unroll
  for (int r = 0; r < 8; ++r) { rmax[r] = -1e30f; rsum[r] = 0.f; }

  const int kend = qi0 + 16;            // causal: last key needed is qi0+15
  for (int ks = 0; ks < kend; ks += 32) {
    // --- scores S = Q K^T (16x32 over two 16x16 wmma tiles, K=64 = 2x32)
    v8f sc[2];
#pragma unroll
    for (int nt = 0; nt < 2; ++nt) {
      v16bf bk0 = load_frag16(Kb + kvb + (size_t)(ks + nt * 16 + ln) * HDIM + koff);
      v16bf bk1 = load_frag16(Kb + kvb + (size_t)(ks + nt * 16 + ln) * HDIM + 32 + koff);
      v8f z = zero_v8f();
      v8f p = __builtin_amdgcn_wmma_f32_16x16x32_bf16(false, aq[0], false, bk0,
                                                      (short)0, z, false, false);
      sc[nt] = __builtin_amdgcn_wmma_f32_16x16x32_bf16(false, aq[1], false, bk1,
                                                       (short)0, p, false, false);
    }

    // --- causal mask + online softmax (half-wave row reductions)
#pragma unroll
    for (int r = 0; r < 8; ++r) {
      const int row = qi0 + rbase + r;
      float vmax = -1e30f;
#pragma unroll
      for (int nt = 0; nt < 2; ++nt) {
        const int col = ks + nt * 16 + ln;
        float xv = sc[nt][r];
        if (col > row) xv = -1e30f;
        sc[nt][r] = xv;
        vmax = fmaxf(vmax, xv);
      }
#pragma unroll
      for (int off = 1; off < 16; off <<= 1)
        vmax = fmaxf(vmax, __shfl_xor(vmax, off, 32));
      const float Mn = fmaxf(rmax[r], vmax);
      const float al = __expf(rmax[r] - Mn);
      rmax[r] = Mn;
      float psum = 0.f;
#pragma unroll
      for (int nt = 0; nt < 2; ++nt) {
        const float pv = __expf(sc[nt][r] - Mn);
        sc[nt][r] = pv;
        psum += pv;
      }
#pragma unroll
      for (int off = 1; off < 16; off <<= 1)
        psum += __shfl_xor(psum, off, 32);
      rsum[r] = rsum[r] * al + psum;
      alpha[r] = al;
    }
#pragma unroll
    for (int d4 = 0; d4 < 4; ++d4)
#pragma unroll
      for (int r = 0; r < 8; ++r) vacc[d4][r] *= alpha[r];

    // --- stage P through wave-private LDS: C layout -> A-fragment layout
#pragma unroll
    for (int nt = 0; nt < 2; ++nt)
#pragma unroll
      for (int r = 0; r < 8; ++r)
        Pds[wid][(rbase + r) * LDT + nt * 16 + ln] = f2bf(sc[nt][r]);
    asm volatile("s_wait_dscnt 0" ::: "memory");
    v16bf ap = load_frag16(&Pds[wid][ln * LDT + koff]);

    // --- O += P V (4 wmma over dim tiles, K = 32 keys)
#pragma unroll
    for (int d4 = 0; d4 < 4; ++d4) {
      v16bf bv;
      const bf16_t* vp = Vb + kvb + (size_t)ks * HDIM + d4 * 16 + ln;
#pragma unroll
      for (int j = 0; j < 8; ++j) {
        bv[j]     = vp[(size_t)(koff + j) * HDIM];
        bv[8 + j] = vp[(size_t)(16 + koff + j) * HDIM];
      }
      vacc[d4] = __builtin_amdgcn_wmma_f32_16x16x32_bf16(false, ap, false, bv,
                                                         (short)0, vacc[d4],
                                                         false, false);
    }
  }

  // --- normalize and store O as [token, h*64+d] bf16 rows for the final GEMM
#pragma unroll
  for (int d4 = 0; d4 < 4; ++d4)
#pragma unroll
    for (int r = 0; r < 8; ++r) {
      const int srow = qi0 + rbase + r;
      const size_t tok = (size_t)b * SEQ + srow;
      O[tok * LATENT + h * HDIM + d4 * 16 + ln] = f2bf(vacc[d4][r] / rsum[r]);
    }
}

// ---------------------------------------------------------------- launcher
extern "C" void kernel_launch(void* const* d_in, const int* in_sizes, int n_in,
                              void* d_out, int out_size, void* d_ws, size_t ws_size,
                              hipStream_t stream) {
  const float* x        = (const float*)d_in[0];
  const float* w_qkv    = (const float*)d_in[1];
  const float* w_o      = (const float*)d_in[2];
  const float* qc1_w    = (const float*)d_in[3];
  const float* qc1_b    = (const float*)d_in[4];
  const float* qc2_w    = (const float*)d_in[5];
  const float* qc2_b    = (const float*)d_in[6];
  const float* kc1_w    = (const float*)d_in[7];
  const float* kc1_b    = (const float*)d_in[8];
  const float* kc2_w    = (const float*)d_in[9];
  const float* kc2_b    = (const float*)d_in[10];
  const float* key_temp = (const float*)d_in[11];
  float* out = (float*)d_out;

  char* wsp = (char*)d_ws;
  size_t off = 0;
  auto alloc = [&](size_t bytes) -> char* {
    char* p = wsp + off;
    off += (bytes + 255) & ~(size_t)255;
    return p;
  };

  bf16_t* xb    = (bf16_t*)alloc((size_t)TOK * DMODEL * 2);       // 33.5 MB
  bf16_t* wqkvb = (bf16_t*)alloc((size_t)QKVOUT * DMODEL * 2);    // 12.6 MB
  bf16_t* wob   = (bf16_t*)alloc((size_t)DMODEL * LATENT * 2);    //  8.4 MB
  float*  qkv   = (float*) alloc((size_t)TOK * QKVOUT * 4);       // 25.2 MB
  float*  q1    = (float*) alloc((size_t)TOK * LATENT * 4);       // 16.8 MB
  float*  k1    = (float*) alloc((size_t)TOK * KVD * 4);          //  4.2 MB
  bf16_t* wcq   = (bf16_t*)alloc((size_t)LATENT * 3 * LATENT * 2);//  6.3 MB
  float*  k2    = (float*) alloc((size_t)TOK * KVD * 4);          //  4.2 MB
  bf16_t* qo    = (bf16_t*)alloc((size_t)TOK * LATENT * 2);       //  8.4 MB
  bf16_t* ko    = (bf16_t*)alloc((size_t)TOK * KVD * 2);          //  2.1 MB
  bf16_t* vo    = (bf16_t*)alloc((size_t)TOK * KVD * 2);          //  2.1 MB
  // aliases: regions dead after the QKV GEMM / tap build (stream-ordered)
  bf16_t* xcq = xb;      // tap-concat matrix [TOK, 3*LATENT] <= x_bf16 size
  float*  q2  = q1;      // qc2 output reuses qc1-output region
  bf16_t* ob  = wqkvb;   // attention output [TOK, LATENT] <= w_qkv_bf16 size

  const int T = 256;
  k_f32_to_bf16<<<(TOK * DMODEL + T - 1) / T, T, 0, stream>>>(x, xb, TOK * DMODEL);
  k_f32_to_bf16<<<(QKVOUT * DMODEL + T - 1) / T, T, 0, stream>>>(w_qkv, wqkvb, QKVOUT * DMODEL);
  k_f32_to_bf16<<<(DMODEL * LATENT + T - 1) / T, T, 0, stream>>>(w_o, wob, DMODEL * LATENT);

  // qkv = x @ w_qkv^T
  k_gemm_bf16_nt<<<dim3(QKVOUT / BN, TOK / BM), T, 0, stream>>>(
      xb, wqkvb, qkv, TOK, QKVOUT, DMODEL);

  // grouped causal conv1 (q: 16 groups, k: 4 groups)
  k_grouped_causal_conv<<<(TOK * LATENT + T - 1) / T, T, 0, stream>>>(
      qkv, QKVOUT, 0, qc1_w, qc1_b, q1, LATENT, HDIM);
  k_grouped_causal_conv<<<(TOK * KVD + T - 1) / T, T, 0, stream>>>(
      qkv, QKVOUT, LATENT, kc1_w, kc1_b, k1, KVD, HDIM);

  // qc2 full conv as GEMM over 3-tap concat (K = 3072)
  k_build_taps<<<(TOK * 3 * LATENT + T - 1) / T, T, 0, stream>>>(q1, xcq, LATENT);
  k_build_wc<<<(LATENT * 3 * LATENT + T - 1) / T, T, 0, stream>>>(qc2_w, wcq, LATENT, LATENT);
  k_gemm_bf16_nt<<<dim3(LATENT / BN, TOK / BM), T, 0, stream>>>(
      xcq, wcq, q2, TOK, LATENT, 3 * LATENT);

  // kc2 full conv (tiny) scalar
  k_kconv2<<<(TOK * KVD + T - 1) / T, T, 0, stream>>>(k1, kc2_w, kc2_b, k2);

  // mixing + l2norm + rope + scale/temperature folds, V shift-concat
  k_finalize_q<<<(TOK * NHEADS * 32 + T - 1) / T, T, 0, stream>>>(qkv, q2, qc2_b, qo);
  k_finalize_kv<<<(TOK * NKV * 32 + T - 1) / T, T, 0, stream>>>(qkv, k2, kc2_b, key_temp, ko, vo);

  // causal flash attention
  k_flash_attn<<<dim3(SEQ / 128, NHEADS, NBATCH), T, 0, stream>>>(qo, ko, vo, ob);

  // out = O @ w_o^T
  k_gemm_bf16_nt<<<dim3(DMODEL / BN, TOK / BM), T, 0, stream>>>(
      ob, wob, out, TOK, DMODEL, LATENT);

  (void)in_sizes; (void)n_in; (void)out_size; (void)ws_size;
}
